// AffinityLoss_8040178778926
// MI455X (gfx1250) — matrix operand order repaired and checked by reading, not compiled, not run on hardware
//
#include <hip/hip_runtime.h>
#include <hip/hip_bf16.h>

typedef __attribute__((ext_vector_type(16))) _Float16 v16h;
typedef __attribute__((ext_vector_type(8)))  _Float16 v8h;
typedef __attribute__((ext_vector_type(8)))  float    v8f;

// ---------------------------------------------------------------------------
// Zero the two fp64 level accumulators in the workspace.
// ---------------------------------------------------------------------------
__global__ void init_acc_kernel(double* acc) {
    if (threadIdx.x < 2) acc[threadIdx.x] = 0.0;
}

// ---------------------------------------------------------------------------
// Convert fp32 r[b][c][hw] -> f16 rT[b][hw][c] (transpose so K (=c) is the
// fast axis: WMMA A/B fragments then load contiguous 16B chunks per lane).
// ---------------------------------------------------------------------------
__global__ void convert_transpose_kernel(const float* __restrict__ in,
                                         _Float16* __restrict__ out,
                                         int C, int HW, int total) {
    int idx = blockIdx.x * blockDim.x + threadIdx.x;
    if (idx >= total) return;
    int i = idx % HW;
    int c = (idx / HW) % C;
    int b = idx / (HW * C);
    out[((size_t)b * HW + i) * C + c] = (_Float16)in[idx];
}

// ---------------------------------------------------------------------------
// Fused affinity-diff kernel over the UPPER-TRIANGULAR 16x16 tiles only
// (Gram matrices are symmetric, so the per-element loss term is symmetric in
// (i,j): off-diagonal tiles are weighted x2 -> 2x less WMMA + softmax work).
//
// One workgroup = 8 waves = one (i,j) tile. Wave w computes the Gram tile for
// (set = w>>2, batch = w&3) with v_wmma_f32_16x16x32_f16, stores to LDS; then
// 256 threads do the 4-way batch softmax for both sets, abs-diff, wave-level
// shfl reduction -> one fp64 atomic per block.
//
// rT layout: [set(2)][batch(4)][HW][C], f16.
// ---------------------------------------------------------------------------
template<int C, int HW>
__global__ void affinity_loss_kernel(const _Float16* __restrict__ rT,
                                     double* __restrict__ acc) {
    constexpr int T = HW / 16;           // tiles per dimension
    __shared__ float gram[8][256];       // [set*4+batch][m*16+n]
    __shared__ float red[8];             // per-wave partials

    const int t    = threadIdx.x;        // 0..255
    const int lane = t & 31;
    const int bs   = t >> 5;             // wave id = set*4 + batch
    const int mn   = lane & 15;          // A: row M; B: col N
    const int hi   = lane >> 4;          // lane half selects K sub-range

    // Map linear block index -> upper-triangular tile (ti <= tj).
    const int k = blockIdx.x;
    int ti = (int)((2.0 * T + 1.0 -
                    sqrt((2.0 * T + 1.0) * (2.0 * T + 1.0) - 8.0 * (double)k)) * 0.5);
    while ((ti + 1) * T - ((ti + 1) * ti) / 2 <= k) ++ti;   // fixups for fp rounding
    while (ti * T - (ti * (ti - 1)) / 2 > k) --ti;
    const int tj = ti + (k - (ti * T - (ti * (ti - 1)) / 2));
    const double w = (ti == tj) ? 1.0 : 2.0;

    const int i0 = ti * 16;
    const int j0 = tj * 16;

    const _Float16* base = rT + (size_t)bs * HW * C;
    const _Float16* rowA = base + (size_t)(i0 + mn) * C;  // A[m][k] = rT[i0+m][k]
    const _Float16* rowB = base + (size_t)(j0 + mn) * C;  // B[k][n] = rT[j0+n][k]

    v8f c8 = {};
    #pragma unroll
    for (int k0 = 0; k0 < C; k0 += 32) {
        v16h a, b;
        // A 16x32 f16 (ISA 7.12.2): lane<16: halves0-7=K0..7, 8-15=K16..23;
        //                           lane>=16: halves0-7=K8..15, 8-15=K24..31.
        *reinterpret_cast<v8h*>(&a)       = *reinterpret_cast<const v8h*>(rowA + k0 + hi * 8);
        *(reinterpret_cast<v8h*>(&a) + 1) = *reinterpret_cast<const v8h*>(rowA + k0 + 16 + hi * 8);
        // B 32x16 f16: lane holds N=mn, same K layout as A (halves0-7=K hi*8.., 8-15=K 16+hi*8..).
        *reinterpret_cast<v8h*>(&b)       = *reinterpret_cast<const v8h*>(rowB + k0 + hi * 8);
        *(reinterpret_cast<v8h*>(&b) + 1) = *reinterpret_cast<const v8h*>(rowB + k0 + 16 + hi * 8);

        c8 = __builtin_amdgcn_wmma_f32_16x16x32_f16(
                 /*neg_a=*/false, a, /*neg_b=*/false, b,
                 /*c_mod=*/(short)0, c8, /*reuse_a=*/false, /*reuse_b=*/false);
    }

    // D layout: VGPR v holds D[m][n], m = v + 8*hi, n = mn.
    #pragma unroll
    for (int v = 0; v < 8; ++v)
        gram[bs][(v + 8 * hi) * 16 + mn] = c8[v];

    __syncthreads();

    // Element t = (m,n). Softmax over the 4-batch axis per set, abs diff.
    float g1[4], g2[4];
    #pragma unroll
    for (int b = 0; b < 4; ++b) { g1[b] = gram[b][t]; g2[b] = gram[4 + b][t]; }

    float m1 = fmaxf(fmaxf(g1[0], g1[1]), fmaxf(g1[2], g1[3]));
    float m2 = fmaxf(fmaxf(g2[0], g2[1]), fmaxf(g2[2], g2[3]));
    float e1[4], e2[4], s1 = 0.f, s2 = 0.f;
    #pragma unroll
    for (int b = 0; b < 4; ++b) {
        e1[b] = __expf(g1[b] - m1); s1 += e1[b];
        e2[b] = __expf(g2[b] - m2); s2 += e2[b];
    }
    float r1 = __builtin_amdgcn_rcpf(s1);   // v_rcp_f32: ~1ulp, no IEEE div seq
    float r2 = __builtin_amdgcn_rcpf(s2);
    float local = 0.f;
    #pragma unroll
    for (int b = 0; b < 4; ++b) local += fabsf(e1[b] * r1 - e2[b] * r2);

    // Wave32 shfl reduction, then one cross-wave pass + one fp64 atomic.
    #pragma unroll
    for (int off = 16; off > 0; off >>= 1) local += __shfl_xor(local, off, 32);
    if (lane == 0) red[bs] = local;
    __syncthreads();
    if (t == 0) {
        double s = 0.0;
        #pragma unroll
        for (int q = 0; q < 8; ++q) s += (double)red[q];
        atomicAdd(acc, w * s);              // global_atomic_add_f64
    }
}

// ---------------------------------------------------------------------------
// loss = 0.5 * (acc0 / (4*4096^2) + acc1 / (4*1024^2))
// ---------------------------------------------------------------------------
__global__ void finalize_kernel(const double* __restrict__ acc,
                                float* __restrict__ out) {
    double l0 = acc[0] / (4.0 * 4096.0 * 4096.0);
    double l1 = acc[1] / (4.0 * 1024.0 * 1024.0);
    out[0] = (float)(0.5 * (l0 + l1));
}

extern "C" void kernel_launch(void* const* d_in, const int* in_sizes, int n_in,
                              void* d_out, int out_size, void* d_ws, size_t ws_size,
                              hipStream_t stream) {
    (void)in_sizes; (void)n_in; (void)out_size; (void)ws_size;

    const float* f1_0 = (const float*)d_in[0];   // [4,64,64,64]
    const float* f1_1 = (const float*)d_in[1];   // [4,128,32,32]
    const float* f2_0 = (const float*)d_in[2];
    const float* f2_1 = (const float*)d_in[3];
    float* out = (float*)d_out;

    // Workspace: [double acc[2]] then f16 rT buffers (6 MB total).
    double*   acc = (double*)d_ws;
    _Float16* h   = (_Float16*)((char*)d_ws + 256);
    const int t0 = 4 * 64 * 4096;    // elements per level-0 tensor
    const int t1 = 4 * 128 * 1024;   // elements per level-1 tensor
    _Float16* h0 = h;                // level 0: [set][b][4096][64]
    _Float16* h1 = h + 2 * t0;       // level 1: [set][b][1024][128]

    init_acc_kernel<<<1, 32, 0, stream>>>(acc);

    convert_transpose_kernel<<<(t0 + 255) / 256, 256, 0, stream>>>(f1_0, h0,      64, 4096, t0);
    convert_transpose_kernel<<<(t0 + 255) / 256, 256, 0, stream>>>(f2_0, h0 + t0, 64, 4096, t0);
    convert_transpose_kernel<<<(t1 + 255) / 256, 256, 0, stream>>>(f1_1, h1,      128, 1024, t1);
    convert_transpose_kernel<<<(t1 + 255) / 256, 256, 0, stream>>>(f2_1, h1 + t1, 128, 1024, t1);

    constexpr int T0 = 4096 / 16, T1 = 1024 / 16;
    affinity_loss_kernel<64, 4096><<<T0 * (T0 + 1) / 2, 256, 0, stream>>>(h0, acc + 0);
    affinity_loss_kernel<128, 1024><<<T1 * (T1 + 1) / 2, 256, 0, stream>>>(h1, acc + 1);

    finalize_kernel<<<1, 1, 0, stream>>>(acc, out);
}
